// CAModel_50414326120608
// MI455X (gfx1250) — compile-verified, matrix operand order checked by reading, not run on hardware
//
#include <hip/hip_runtime.h>
#include <hip/hip_bf16.h>
#include <hip/hip_fp16.h>

typedef __attribute__((ext_vector_type(16))) _Float16 v16h;
typedef __attribute__((ext_vector_type(8)))  _Float16 v8h;
typedef __attribute__((ext_vector_type(8)))  float    v8f;

// Geometry
static constexpr int Bn = 64, Hn = 64, Wn = 64, Cn = 8;
static constexpr int PIX = Bn * Hn * Wn;            // 262144
static constexpr int M1  = Bn * 62 * 62;            // 246016 (divisible by 32)
// GEMM shapes (K padded to multiple of 32)
static constexpr int K_C1 = 648, KT_C1 = 21, NT_C1 = 3;   // conv1: 9*72 -> 48
static constexpr int K_CT = 432, KT_CT = 14, NT_CT = 3;   // convT: 9*48 -> 48
static constexpr int K_F1 = 48,  KT_F1 = 2,  NT_F1 = 8;   // 1x1:   48 -> 128
static constexpr int K_F2 = 128, KT_F2 = 4,  NT_F2 = 1;   // 1x1:   128 -> 8 (N padded to 16)

#define WMMA_F16(A, Bf, C) \
    __builtin_amdgcn_wmma_f32_16x16x32_f16(false, (A), false, (Bf), (short)0, (C), false, false)

// ---- WMMA f16 fragment index helpers (wave32, 16x16x32) ----
// A/B 16-bit fragment: lane L holds row (L&15); element e maps to
// K = (e&7) + 16*(e>>3) + 8*(L>=16) => two contiguous 8-element runs
// at K0 = kt*32 + laneHi*8 and K0+16.
__device__ __forceinline__ int fragK(int e, int laneHi) {
    return (e & 7) | ((e >> 3) << 4) | (laneHi << 3);
}
__device__ __forceinline__ v16h combine(v8h lo, v8h hi) {
    return __builtin_shufflevector(lo, hi, 0, 1, 2, 3, 4, 5, 6, 7,
                                          8, 9, 10, 11, 12, 13, 14, 15);
}

// conv1 im2col run: K = ky*216 + kx*72 + c; memory offset (halves) relative to
// the output pixel base is k8 + (k8/216)*4392   (64*72 - 216 = 4392).
__device__ __forceinline__ v8h run_c1_nc(const _Float16* __restrict__ pix, int k8) {
    int ky = k8 / 216;
    return *(const v8h*)(pix + k8 + ky * 4392);
}
__device__ __forceinline__ v8h run_c1_g(const _Float16* __restrict__ pix, int k8) {
    v8h z = {};
    if (k8 >= K_C1) return z;
    return run_c1_nc(pix, k8);
}

// convT im2col run: K = ky*144 + kx*48 + ci; pt0 is &h1[(b,yy,xx) row];
// offset = c0 - ky*2976 - kx*48 with spatial validity (yy-ky, xx-kx) in [0,62).
__device__ __forceinline__ v8h run_ct(const _Float16* __restrict__ pt0,
                                      int yy, int xx, int k8) {
    v8h z = {};
    if (k8 >= K_CT) return z;
    int ky = k8 / 144;
    int r  = k8 - ky * 144;
    int kx = r / 48;
    int c0 = r - kx * 48;
    int iy = yy - ky, ix = xx - kx;
    if (iy < 0 || iy >= 62 || ix < 0 || ix >= 62) return z;
    return *(const v8h*)(pt0 + c0 - ky * 2976 - kx * 48);
}

// ---------------------------------------------------------------------------
// 0) init: y_sum = b_att, sumsq = 0
__global__ void init_ysum(const float* __restrict__ b_att, float* __restrict__ ysum,
                          float* __restrict__ sumsq) {
    int i = blockIdx.x * blockDim.x + threadIdx.x;
    if (i < 648) ysum[i] = b_att[i];
    if (i == 0)  sumsq[0] = 0.f;
}

// ---------------------------------------------------------------------------
// 1) pack conv weights [K][N] fp32 -> WMMA B-fragment-major f16 (zero padded)
__global__ void pack_b(const float* __restrict__ Wsrc, _Float16* __restrict__ out,
                       int K, int N, int KT, int NT) {
    int tid = blockIdx.x * blockDim.x + threadIdx.x;
    int total = KT * NT * 32 * 16;
    if (tid >= total) return;
    int e    = tid & 15;
    int lane = (tid >> 4) & 31;
    int t    = tid >> 9;            // kt*NT + nt
    int nt   = t % NT;
    int kt   = t / NT;
    int k    = kt * 32 + fragK(e, lane >> 4);
    int n    = nt * 16 + (lane & 15);
    float v  = (k < K && n < N) ? Wsrc[(size_t)k * N + n] : 0.f;
    out[tid] = (_Float16)v;
}

// ---------------------------------------------------------------------------
// 2) perceive: fixed stencil (ident / corner / corner), depthwise, SAME zero-pad
__global__ __launch_bounds__(256) void perceive(const float* __restrict__ x,
                                                float* __restrict__ y) {
    int m = blockIdx.x * blockDim.x + threadIdx.x;       // pixel id
    int b = m >> 12, yy = (m >> 6) & 63, xx = m & 63;
    #pragma unroll
    for (int g = 0; g < Cn; ++g) {
        float center = x[(size_t)m * Cn + g];
        float corners = 0.f;
        #pragma unroll
        for (int dy = -1; dy <= 1; dy += 2) {
            int ny = yy + dy; if (ny < 0 || ny > 63) continue;
            #pragma unroll
            for (int dx = -1; dx <= 1; dx += 2) {
                int nx = xx + dx; if (nx < 0 || nx > 63) continue;
                corners += x[((size_t)((b << 12) + ny * 64 + nx)) * Cn + g];
            }
        }
        float d = corners * 0.125f;                      // dxk == dyk (symmetric)
        y[(size_t)m * 24 + g * 3 + 0] = center;
        y[(size_t)m * 24 + g * 3 + 1] = d;
        y[(size_t)m * 24 + g * 3 + 2] = d;
    }
}

// ---------------------------------------------------------------------------
// 3) attention matvec: y_sum += y[0].flat (98304) @ W_att (98304x648).
//    Streams the 255 MB matrix fully coalesced; register partials + atomics.
__global__ __launch_bounds__(256) void attn_matvec(const float* __restrict__ yv,
                                                   const float* __restrict__ Watt,
                                                   float* __restrict__ ysum) {
    int tid = threadIdx.x;
    int c0 = tid, c1 = tid + 256, c2 = tid + 512;        // c2 valid iff < 648
    float a0 = 0.f, a1 = 0.f, a2 = 0.f;
    int r0 = blockIdx.x * 128;
    for (int rr = 0; rr < 128; ++rr) {
        int r = r0 + rr;
        float yval = yv[r];
        const float* Wr = Watt + (size_t)r * 648;
        a0 += yval * Wr[c0];
        a1 += yval * Wr[c1];
        if (c2 < 648) a2 += yval * Wr[c2];
    }
    atomicAdd(&ysum[c0], a0);
    atomicAdd(&ysum[c1], a1);
    if (c2 < 648) atomicAdd(&ysum[c2], a2);
}

// ---------------------------------------------------------------------------
// 4) second depthwise conv with the learned y_sum kernel, write un-normalized
//    f16 activations, and reduce global sum(y2*y2) for the L2 normalizer.
__global__ __launch_bounds__(256) void dwconv2(const float* __restrict__ y,
                                               const float* __restrict__ ysum,
                                               _Float16* __restrict__ y2h,
                                               float* __restrict__ sumsq) {
    __shared__ float ks[648];
    __shared__ float red[256];
    for (int i = threadIdx.x; i < 648; i += 256) ks[i] = ysum[i];
    __syncthreads();
    int m = blockIdx.x * 256 + threadIdx.x;
    int b = m >> 12, yy = (m >> 6) & 63, xx = m & 63;
    float acc[72];
    #pragma unroll
    for (int o = 0; o < 72; ++o) acc[o] = 0.f;
    for (int ky = 0; ky < 3; ++ky) {
        int iy = yy + ky - 1; if (iy < 0 || iy > 63) continue;
        for (int kx = 0; kx < 3; ++kx) {
            int ix = xx + kx - 1; if (ix < 0 || ix > 63) continue;
            const float* yp = y + ((size_t)((b << 12) + iy * 64 + ix)) * 24;
            int kb = (ky * 3 + kx) * 72;                 // ((ky*3+kx)*24+g)*3+j
            #pragma unroll
            for (int g = 0; g < 24; ++g) {
                float v = yp[g];
                acc[g * 3 + 0] += v * ks[kb + g * 3 + 0];
                acc[g * 3 + 1] += v * ks[kb + g * 3 + 1];
                acc[g * 3 + 2] += v * ks[kb + g * 3 + 2];
            }
        }
    }
    float local = 0.f;
    #pragma unroll
    for (int o = 0; o < 72; ++o) {
        float v = acc[o];
        local += v * v;
        y2h[(size_t)m * 72 + o] = (_Float16)v;
    }
    red[threadIdx.x] = local;
    __syncthreads();
    for (int st = 128; st > 0; st >>= 1) {
        if (threadIdx.x < st) red[threadIdx.x] += red[threadIdx.x + st];
        __syncthreads();
    }
    if (threadIdx.x == 0) atomicAdd(sumsq, red[0]);
}

// ---------------------------------------------------------------------------
// 5) conv1: 3x3 VALID, 72->48, im2col WMMA GEMM.
//    2 M-tiles x 3 N-tiles per wave; L2-norm scale folded into epilogue.
__global__ __launch_bounds__(128) void conv1_wmma(const _Float16* __restrict__ y2h,
                                                  const _Float16* __restrict__ Bp,
                                                  const float* __restrict__ b1,
                                                  const float* __restrict__ sumsq,
                                                  _Float16* __restrict__ h1) {
    const int wave = threadIdx.x >> 5;
    const int lane = threadIdx.x & 31;
    const int laneHi = lane >> 4;
    const int pair = blockIdx.x * 4 + wave;             // two m-tiles per wave
    const int m0 = pair * 32 + (lane & 15);
    const int m1 = m0 + 16;
    const float s = rsqrtf(fmaxf(sumsq[0], 1e-12f));

    auto pixptr = [&](int m) -> const _Float16* {
        int b = m / 3844;
        int rm = m - b * 3844;
        int oy = rm / 62;
        int ox = rm - oy * 62;
        return y2h + ((size_t)((b << 12) + oy * 64 + ox)) * 72;
    };
    const _Float16* pix0 = pixptr(m0);
    const _Float16* pix1 = pixptr(m1);

    v8f acc[6];
    v8f zf = {};
    #pragma unroll
    for (int i = 0; i < 6; ++i) acc[i] = zf;

    for (int kt = 0; kt < KT_C1 - 1; ++kt) {            // guard-free main loop
        int k8 = kt * 32 + laneHi * 8;
        v16h a0 = combine(run_c1_nc(pix0, k8), run_c1_nc(pix0, k8 + 16));
        v16h a1 = combine(run_c1_nc(pix1, k8), run_c1_nc(pix1, k8 + 16));
        const _Float16* bp = Bp + (size_t)kt * (NT_C1 * 512) + (size_t)lane * 16;
        v16h bf0 = *(const v16h*)(bp);
        v16h bf1 = *(const v16h*)(bp + 512);
        v16h bf2 = *(const v16h*)(bp + 1024);
        acc[0] = WMMA_F16(a0, bf0, acc[0]);
        acc[1] = WMMA_F16(a0, bf1, acc[1]);
        acc[2] = WMMA_F16(a0, bf2, acc[2]);
        acc[3] = WMMA_F16(a1, bf0, acc[3]);
        acc[4] = WMMA_F16(a1, bf1, acc[4]);
        acc[5] = WMMA_F16(a1, bf2, acc[5]);
    }
    {                                                   // kt = 20: K-pad guard
        const int kt = KT_C1 - 1;
        int k8 = kt * 32 + laneHi * 8;
        v16h a0 = combine(run_c1_g(pix0, k8), run_c1_g(pix0, k8 + 16));
        v16h a1 = combine(run_c1_g(pix1, k8), run_c1_g(pix1, k8 + 16));
        const _Float16* bp = Bp + (size_t)kt * (NT_C1 * 512) + (size_t)lane * 16;
        v16h bf0 = *(const v16h*)(bp);
        v16h bf1 = *(const v16h*)(bp + 512);
        v16h bf2 = *(const v16h*)(bp + 1024);
        acc[0] = WMMA_F16(a0, bf0, acc[0]);
        acc[1] = WMMA_F16(a0, bf1, acc[1]);
        acc[2] = WMMA_F16(a0, bf2, acc[2]);
        acc[3] = WMMA_F16(a1, bf0, acc[3]);
        acc[4] = WMMA_F16(a1, bf1, acc[4]);
        acc[5] = WMMA_F16(a1, bf2, acc[5]);
    }
    const int col = lane & 15;
    const float bb0 = b1[col], bb1 = b1[col + 16], bb2 = b1[col + 32];
    #pragma unroll
    for (int t = 0; t < 2; ++t) {
        #pragma unroll
        for (int r = 0; r < 8; ++r) {
            size_t mo = (size_t)(pair * 32 + t * 16 + r + 8 * laneHi) * 48 + col;
            h1[mo]      = (_Float16)(s * acc[t * 3 + 0][r] + bb0);
            h1[mo + 16] = (_Float16)(s * acc[t * 3 + 1][r] + bb1);
            h1[mo + 32] = (_Float16)(s * acc[t * 3 + 2][r] + bb2);
        }
    }
}

// ---------------------------------------------------------------------------
// 6) conv-transpose 3x3 VALID 48->48: out[y,x] = sum h[y-ky,x-kx]*Wt[ky,kx]
//    2 M-tiles x 3 N-tiles per wave; zero gather OOB.
__global__ __launch_bounds__(128) void convT_wmma(const _Float16* __restrict__ h1,
                                                  const _Float16* __restrict__ Bp,
                                                  const float* __restrict__ bt,
                                                  _Float16* __restrict__ h2) {
    const int wave = threadIdx.x >> 5;
    const int lane = threadIdx.x & 31;
    const int laneHi = lane >> 4;
    const int pair = blockIdx.x * 4 + wave;
    const int m0 = pair * 32 + (lane & 15);
    const int m1 = m0 + 16;
    const int b0 = m0 >> 12, y0 = (m0 >> 6) & 63, x0 = m0 & 63;
    const int b1i = m1 >> 12, y1 = (m1 >> 6) & 63, x1 = m1 & 63;
    const _Float16* pt0 = h1 + (size_t)(b0 * 3844 + y0 * 62 + x0) * 48;
    const _Float16* pt1 = h1 + (size_t)(b1i * 3844 + y1 * 62 + x1) * 48;

    v8f acc[6];
    v8f zf = {};
    #pragma unroll
    for (int i = 0; i < 6; ++i) acc[i] = zf;

    for (int kt = 0; kt < KT_CT; ++kt) {
        int k8 = kt * 32 + laneHi * 8;
        v16h a0 = combine(run_ct(pt0, y0, x0, k8), run_ct(pt0, y0, x0, k8 + 16));
        v16h a1 = combine(run_ct(pt1, y1, x1, k8), run_ct(pt1, y1, x1, k8 + 16));
        const _Float16* bp = Bp + (size_t)kt * (NT_CT * 512) + (size_t)lane * 16;
        v16h bf0 = *(const v16h*)(bp);
        v16h bf1 = *(const v16h*)(bp + 512);
        v16h bf2 = *(const v16h*)(bp + 1024);
        acc[0] = WMMA_F16(a0, bf0, acc[0]);
        acc[1] = WMMA_F16(a0, bf1, acc[1]);
        acc[2] = WMMA_F16(a0, bf2, acc[2]);
        acc[3] = WMMA_F16(a1, bf0, acc[3]);
        acc[4] = WMMA_F16(a1, bf1, acc[4]);
        acc[5] = WMMA_F16(a1, bf2, acc[5]);
    }
    const int col = lane & 15;
    const float bb0 = bt[col], bb1 = bt[col + 16], bb2 = bt[col + 32];
    #pragma unroll
    for (int t = 0; t < 2; ++t) {
        #pragma unroll
        for (int r = 0; r < 8; ++r) {
            size_t mo = (size_t)(pair * 32 + t * 16 + r + 8 * laneHi) * 48 + col;
            h2[mo]      = (_Float16)(acc[t * 3 + 0][r] + bb0);
            h2[mo + 16] = (_Float16)(acc[t * 3 + 1][r] + bb1);
            h2[mo + 32] = (_Float16)(acc[t * 3 + 2][r] + bb2);
        }
    }
}

// ---------------------------------------------------------------------------
// 7) 1x1 conv 48->128 + ReLU.  2 M-tiles x 4 N-tiles per wave, gridDim.y = 2
//    selects the N half.  (M=262144, K=48->64, N=128)
__global__ __launch_bounds__(128) void fc1_wmma(const _Float16* __restrict__ h2,
                                                const _Float16* __restrict__ Bp,
                                                const float* __restrict__ b2,
                                                _Float16* __restrict__ h3) {
    const int wave = threadIdx.x >> 5;
    const int lane = threadIdx.x & 31;
    const int laneHi = lane >> 4;
    const int pair = blockIdx.x * 4 + wave;
    const int ntBase = blockIdx.y * 4;
    const int m0 = pair * 32 + (lane & 15);
    const int m1 = m0 + 16;
    const _Float16* row0 = h2 + (size_t)m0 * 48;
    const _Float16* row1 = h2 + (size_t)m1 * 48;

    v8f acc[8];
    v8f zf = {};
    #pragma unroll
    for (int i = 0; i < 8; ++i) acc[i] = zf;
    v8h z8 = {};
    #pragma unroll
    for (int kt = 0; kt < KT_F1; ++kt) {
        int k8 = kt * 32 + laneHi * 8;
        bool vlo = (k8 < K_F1), vhi = (k8 + 16 < K_F1);
        v16h a0 = combine(vlo ? *(const v8h*)(row0 + k8) : z8,
                          vhi ? *(const v8h*)(row0 + k8 + 16) : z8);
        v16h a1 = combine(vlo ? *(const v8h*)(row1 + k8) : z8,
                          vhi ? *(const v8h*)(row1 + k8 + 16) : z8);
        #pragma unroll
        for (int nt = 0; nt < 4; ++nt) {
            v16h bf = *(const v16h*)(Bp + ((size_t)(kt * NT_F1 + ntBase + nt) * 32 + lane) * 16);
            acc[nt]     = WMMA_F16(a0, bf, acc[nt]);
            acc[4 + nt] = WMMA_F16(a1, bf, acc[4 + nt]);
        }
    }
    #pragma unroll
    for (int nt = 0; nt < 4; ++nt) {
        int col = (ntBase + nt) * 16 + (lane & 15);
        float bias = b2[col];
        #pragma unroll
        for (int t = 0; t < 2; ++t) {
            #pragma unroll
            for (int r = 0; r < 8; ++r) {
                size_t mo = (size_t)(pair * 32 + t * 16 + r + 8 * laneHi) * 128 + col;
                h3[mo] = (_Float16)fmaxf(acc[t * 4 + nt][r] + bias, 0.f);
            }
        }
    }
}

// ---------------------------------------------------------------------------
// 8) 1x1 conv 128->8  (K=128, N=8 padded to 16), 2 M-tiles per wave, fp32 out
__global__ __launch_bounds__(128) void fc2_wmma(const _Float16* __restrict__ h3,
                                                const _Float16* __restrict__ Bp,
                                                const float* __restrict__ b3,
                                                float* __restrict__ dxo) {
    const int wave = threadIdx.x >> 5;
    const int lane = threadIdx.x & 31;
    const int laneHi = lane >> 4;
    const int pair = blockIdx.x * 4 + wave;
    const int m0 = pair * 32 + (lane & 15);
    const int m1 = m0 + 16;
    const _Float16* row0 = h3 + (size_t)m0 * 128;
    const _Float16* row1 = h3 + (size_t)m1 * 128;
    v8f acc0 = {}, acc1 = {};
    #pragma unroll
    for (int kt = 0; kt < KT_F2; ++kt) {
        int k8 = kt * 32 + laneHi * 8;
        v16h a0 = combine(*(const v8h*)(row0 + k8), *(const v8h*)(row0 + k8 + 16));
        v16h a1 = combine(*(const v8h*)(row1 + k8), *(const v8h*)(row1 + k8 + 16));
        v16h bf = *(const v16h*)(Bp + ((size_t)kt * 32 + lane) * 16);
        acc0 = WMMA_F16(a0, bf, acc0);
        acc1 = WMMA_F16(a1, bf, acc1);
    }
    const int col = lane & 15;
    if (col < 8) {
        float bias = b3[col];
        #pragma unroll
        for (int r = 0; r < 8; ++r) {
            size_t mo0 = (size_t)(pair * 32 + r + 8 * laneHi) * 8 + col;
            size_t mo1 = (size_t)(pair * 32 + 16 + r + 8 * laneHi) * 8 + col;
            dxo[mo0] = acc0[r] + bias;
            dxo[mo1] = acc1[r] + bias;
        }
    }
}

// ---------------------------------------------------------------------------
// 9) tail: x2 = x + dx*mask ; life = (maxpool3(x.a)>0.1) & (maxpool3(x2.a)>0.1)
__global__ __launch_bounds__(256) void finalize(const float* __restrict__ x,
                                                const float* __restrict__ noise,
                                                const float* __restrict__ dxo,
                                                float* __restrict__ out) {
    int m = blockIdx.x * 256 + threadIdx.x;
    int b = m >> 12, yy = (m >> 6) & 63, xx = m & 63;
    float mask = (noise[m] <= 0.5f) ? 1.f : 0.f;
    float pre = -1e30f, post = -1e30f;
    for (int dy = -1; dy <= 1; ++dy) {
        int ny = yy + dy; if (ny < 0 || ny > 63) continue;
        for (int dx = -1; dx <= 1; ++dx) {
            int nx = xx + dx; if (nx < 0 || nx > 63) continue;
            int nm = (b << 12) + ny * 64 + nx;
            float xa = x[(size_t)nm * Cn + 3];
            pre = fmaxf(pre, xa);
            float nmask = (noise[nm] <= 0.5f) ? 1.f : 0.f;
            post = fmaxf(post, xa + dxo[(size_t)nm * Cn + 3] * nmask);
        }
    }
    float life = (pre > 0.1f && post > 0.1f) ? 1.f : 0.f;
    #pragma unroll
    for (int c = 0; c < Cn; ++c) {
        float x2 = x[(size_t)m * Cn + c] + dxo[(size_t)m * Cn + c] * mask;
        out[(size_t)m * Cn + c] = x2 * life;
    }
}

// ---------------------------------------------------------------------------
extern "C" void kernel_launch(void* const* d_in, const int* in_sizes, int n_in,
                              void* d_out, int out_size, void* d_ws, size_t ws_size,
                              hipStream_t stream) {
    (void)in_sizes; (void)n_in; (void)out_size; (void)ws_size;
    const float* x     = (const float*)d_in[0];
    const float* noise = (const float*)d_in[1];
    const float* Watt  = (const float*)d_in[2];
    const float* b_att = (const float*)d_in[3];
    const float* W1    = (const float*)d_in[4];
    const float* b1    = (const float*)d_in[5];
    const float* Wt    = (const float*)d_in[6];
    const float* bt    = (const float*)d_in[7];
    const float* W2    = (const float*)d_in[8];
    const float* b2    = (const float*)d_in[9];
    const float* W3    = (const float*)d_in[10];
    const float* b3    = (const float*)d_in[11];
    float* out = (float*)d_out;

    // workspace layout (256B aligned)
    char* ws = (char*)d_ws;
    size_t off = 0;
    auto take = [&](size_t bytes) -> void* {
        void* p = ws + off;
        off = (off + bytes + 255) & ~(size_t)255;
        return p;
    };
    float*    ysum  = (float*)take(648 * 4);
    float*    sumsq = (float*)take(4);
    float*    ybuf  = (float*)take((size_t)PIX * 24 * 4);
    _Float16* y2h   = (_Float16*)take((size_t)PIX * 72 * 2);
    _Float16* h1    = (_Float16*)take((size_t)M1  * 48 * 2);
    _Float16* h2    = (_Float16*)take((size_t)PIX * 48 * 2);
    _Float16* h3    = (_Float16*)take((size_t)PIX * 128 * 2);
    float*    dxo   = (float*)take((size_t)PIX * 8 * 4);
    _Float16* W1p   = (_Float16*)take((size_t)KT_C1 * NT_C1 * 512 * 2);
    _Float16* Wtp   = (_Float16*)take((size_t)KT_CT * NT_CT * 512 * 2);
    _Float16* W2p   = (_Float16*)take((size_t)KT_F1 * NT_F1 * 512 * 2);
    _Float16* W3p   = (_Float16*)take((size_t)KT_F2 * NT_F2 * 512 * 2);

    // 0) init y_sum with bias, zero sumsq
    init_ysum<<<3, 256, 0, stream>>>(b_att, ysum, sumsq);
    // 1) pack all conv weights into WMMA fragment layout
    pack_b<<<(KT_C1 * NT_C1 * 512 + 255) / 256, 256, 0, stream>>>(W1, W1p, K_C1, 48, KT_C1, NT_C1);
    pack_b<<<(KT_CT * NT_CT * 512 + 255) / 256, 256, 0, stream>>>(Wt, Wtp, K_CT, 48, KT_CT, NT_CT);
    pack_b<<<(KT_F1 * NT_F1 * 512 + 255) / 256, 256, 0, stream>>>(W2, W2p, K_F1, 128, KT_F1, NT_F1);
    pack_b<<<(KT_F2 * NT_F2 * 512 + 255) / 256, 256, 0, stream>>>(W3, W3p, K_F2, 8, KT_F2, NT_F2);
    // 2) perceive stencil
    perceive<<<PIX / 256, 256, 0, stream>>>(x, ybuf);
    // 3) attention matvec over sample 0 (streams the 255 MB matrix once)
    attn_matvec<<<98304 / 128, 256, 0, stream>>>(ybuf, Watt, ysum);
    // 4) second depthwise conv + global sum of squares
    dwconv2<<<PIX / 256, 256, 0, stream>>>(ybuf, ysum, y2h, sumsq);
    // 5..8) WMMA GEMM stack (2 M-tiles per wave, 4 waves per block)
    conv1_wmma<<<(M1 / 32) / 4, 128, 0, stream>>>(y2h, W1p, b1, sumsq, h1);
    convT_wmma<<<(PIX / 32) / 4, 128, 0, stream>>>(h1, Wtp, bt, h2);
    fc1_wmma<<<dim3((PIX / 32) / 4, 2), 128, 0, stream>>>(h2, W2p, b2, h3);
    fc2_wmma<<<(PIX / 32) / 4, 128, 0, stream>>>(h3, W3p, b3, dxo);
    // 9) stochastic update + alive mask
    finalize<<<PIX / 256, 256, 0, stream>>>(x, noise, dxo, out);
}
